// TimeShiftedRecurrentLayer_43550968381898
// MI455X (gfx1250) — compile-verified, matrix operand order checked by reading, not run on hardware
//
#include <hip/hip_runtime.h>

// Bidirectional 3-layer LSTM (AWD-LSTM style with AR/TAR) for MI455X (gfx1250).
// Strategy:
//  - bf16 WMMA (v_wmma_f32_16x16x32_bf16) for both the batched input-projection
//    GEMM and the per-step recurrent GEMM, fp32 accumulate.
//  - One persistent scan kernel per (layer,direction): 32 WGs, global step
//    barrier via per-step arrival counters (all 6 Whh bf16 = 48MB sit in L2).
//  - Cell state c and TAR's h_{t-1} live in registers (fixed thread<->(m,h) map).
//  - Last-layer scan fuses AR/TAR partial sums and writes the time-shifted
//    concat output directly into d_out.
// Workspace: ~303 MB.

typedef __bf16 bf16_t;
typedef __attribute__((ext_vector_type(16))) __bf16 v16bf;
typedef __attribute__((ext_vector_type(8)))  float  v8f;

#define HID 1024
#define G4  4096
#define TT  512   // output timesteps
#define TS  513   // T+1 (zero-prepended)
#define NB  32    // batch
#define NWG 32    // workgroups in persistent scan kernel

static __device__ __forceinline__ v8f wmma_bf16(v16bf a, v16bf b, v8f c) {
  // (neg_a, A, neg_b, B, c_mod, C, reuse_a, reuse_b)
  return __builtin_amdgcn_wmma_f32_16x16x32_bf16(false, a, false, b, (short)0, c,
                                                 false, false);
}

// ---------------------------------------------------------------- utilities
__global__ void f32_to_bf16_kernel(const float* __restrict__ src,
                                   bf16_t* __restrict__ dst, int n) {
  int i = blockIdx.x * blockDim.x + threadIdx.x;
  if (i < n) dst[i] = (bf16_t)src[i];
}

__global__ void combine_bias_kernel(const float* __restrict__ bih,
                                    const float* __restrict__ bhh,
                                    float* __restrict__ dst, int n) {
  int i = blockIdx.x * blockDim.x + threadIdx.x;
  if (i < n) dst[i] = bih[i] + bhh[i];
}

// Build zero-prepended (and optionally time-reversed) bf16 input sequence.
__global__ void prep_seq_kernel(const float* __restrict__ x,
                                bf16_t* __restrict__ seq, int reverse) {
  int i = blockIdx.x * blockDim.x + threadIdx.x;
  const int total = TS * NB * HID;
  if (i >= total) return;
  int t = i / (NB * HID);
  int rem = i - t * (NB * HID);
  float v = 0.0f;
  if (t > 0) {
    int ts = t - 1;
    if (reverse) ts = TT - 1 - ts;
    v = x[(size_t)ts * (NB * HID) + rem];
  }
  seq[i] = (bf16_t)v;
}

__global__ void zero_counters_kernel(int* __restrict__ c, int n) {
  int i = blockIdx.x * blockDim.x + threadIdx.x;
  if (i < n) c[i] = 0;
}

// --------------------------------------------------- batched input projection
// XG[t,m,:] = seq[t,m,:] @ Wih^T + (bih+bhh),  stored bf16.
// grid = (513, 16); block = 256 (8 waves). WG tile 32(M) x 256(N); wave 32x32.
__global__ __launch_bounds__(256) void gemm_xg_kernel(
    const bf16_t* __restrict__ seq, const bf16_t* __restrict__ W,
    const float* __restrict__ bias, bf16_t* __restrict__ xg) {
  const int t    = blockIdx.x;            // timestep (32 rows)
  const int nb   = blockIdx.y;            // 256-col block
  const int wave = threadIdx.x >> 5;
  const int lane = threadIdx.x & 31;
  const int n0   = nb * 256 + wave * 32;
  const int mrow = t * NB;
  const int ar = lane & 15;               // row-in-tile
  const int kh = (lane >> 4) * 16;        // K half selected by upper lanes

  v8f c00 = {}, c01 = {}, c10 = {}, c11 = {};
  const bf16_t* A0 = seq + ((size_t)(mrow + ar)) * HID + kh;
  const bf16_t* A1 = A0 + (size_t)16 * HID;
  const bf16_t* B0 = W + ((size_t)(n0 + ar)) * HID + kh;
  const bf16_t* B1 = B0 + (size_t)16 * HID;

  for (int k = 0; k < HID; k += 32) {
    __builtin_prefetch(B0 + k + 256, 0, 3);   // global_prefetch_b8
    v16bf a0 = *(const v16bf*)(A0 + k);
    v16bf a1 = *(const v16bf*)(A1 + k);
    v16bf b0 = *(const v16bf*)(B0 + k);
    v16bf b1 = *(const v16bf*)(B1 + k);
    c00 = wmma_bf16(a0, b0, c00);
    c01 = wmma_bf16(a0, b1, c01);
    c10 = wmma_bf16(a1, b0, c10);
    c11 = wmma_bf16(a1, b1, c11);
  }

  const int col = lane & 15;
  const int mb  = (lane >> 4) * 8;        // C layout: VGPR r -> M=r (lo) / r+8 (hi)
  const float bias0 = bias[n0 + col];
  const float bias1 = bias[n0 + 16 + col];
  bf16_t* xr = xg + (size_t)mrow * G4;
#pragma unroll
  for (int r = 0; r < 8; ++r) {
    int m0 = mb + r, m1 = 16 + mb + r;
    xr[(size_t)m0 * G4 + n0 + col]      = (bf16_t)(c00[r] + bias0);
    xr[(size_t)m0 * G4 + n0 + 16 + col] = (bf16_t)(c01[r] + bias1);
    xr[(size_t)m1 * G4 + n0 + col]      = (bf16_t)(c10[r] + bias0);
    xr[(size_t)m1 * G4 + n0 + 16 + col] = (bf16_t)(c11[r] + bias1);
  }
}

// ------------------------------------------------------- persistent LSTM scan
// grid = 32 WGs x 256 threads. WG wg owns hidden cols [wg*32, wg*32+32).
// Per step: 8 waves compute the 4 gate tiles (h_{t-1} @ Whh^T) via WMMA,
// exchange through LDS, 256 threads apply the cell update (c, h_{t-1} in
// registers), write h_t as bf16 into seqout (recurrence + next layer input),
// then all 32 WGs meet at a per-step global arrival counter.
__global__ __launch_bounds__(256) void lstm_scan_kernel(
    const bf16_t* __restrict__ xg, const bf16_t* __restrict__ Whh,
    bf16_t* __restrict__ seqout, int* __restrict__ counters,
    float* __restrict__ out, int is_last, int dir,
    float* __restrict__ alphaP, float* __restrict__ betaP) {
  __shared__ float gsh[4][NB][32];   // 16 KB gate exchange
  __shared__ float red[256];

  const int wg    = blockIdx.x;
  const int hbase = wg * 32;
  const int tid   = threadIdx.x;
  const int wave  = tid >> 5;
  const int lane  = tid & 31;
  const int q     = wave >> 1;             // gate: 0=i 1=f 2=g 3=o
  const int off   = (wave & 1) * 16;
  const int ncol  = q * HID + hbase + off; // gate-row tile base in Whh
  const int ar    = lane & 15;
  const int kh    = (lane >> 4) * 16;
  const int col   = lane & 15;
  const int mb    = (lane >> 4) * 8;

  float creg[4], prevh[4];
#pragma unroll
  for (int j = 0; j < 4; ++j) { creg[j] = 0.0f; prevh[j] = 0.0f; }
  float a_acc = 0.0f, b_acc = 0.0f;

  const bf16_t* Brow = Whh + ((size_t)(ncol + ar)) * HID + kh;

  for (int t = 0; t < TS; ++t) {
    v8f r0 = {}, r1 = {};
    if (t > 0) {
      const bf16_t* A0 = seqout + (((size_t)(t - 1)) * NB + ar) * HID + kh;
      const bf16_t* A1 = A0 + (size_t)16 * HID;
      for (int k = 0; k < HID; k += 32) {
        v16bf a0 = *(const v16bf*)(A0 + k);
        v16bf a1 = *(const v16bf*)(A1 + k);
        v16bf b  = *(const v16bf*)(Brow + k);
        r0 = wmma_bf16(a0, b, r0);
        r1 = wmma_bf16(a1, b, r1);
      }
    }
#pragma unroll
    for (int r = 0; r < 8; ++r) {
      gsh[q][mb + r][off + col]      = r0[r];
      gsh[q][16 + mb + r][off + col] = r1[r];
    }
    __syncthreads();

#pragma unroll
    for (int j = 0; j < 4; ++j) {
      int flat = tid + j * 256;            // fixed (m,hh) per thread across t
      int m = flat >> 5, hh = flat & 31;
      size_t xb = (((size_t)t) * NB + m) * G4 + hbase + hh;
      float gi = gsh[0][m][hh] + (float)xg[xb];
      float gf = gsh[1][m][hh] + (float)xg[xb + HID];
      float gg = gsh[2][m][hh] + (float)xg[xb + 2 * HID];
      float go = gsh[3][m][hh] + (float)xg[xb + 3 * HID];
      float si = 1.0f / (1.0f + __expf(-gi));
      float sf = 1.0f / (1.0f + __expf(-gf));
      float so = 1.0f / (1.0f + __expf(-go));
      float tg = tanhf(gg);
      float cn = sf * creg[j] + si * tg;
      creg[j] = cn;
      float hn = so * tanhf(cn);
      seqout[(((size_t)t) * NB + m) * HID + hbase + hh] = (bf16_t)hn;
      if (is_last) {
        a_acc += hn * hn;                                   // AR
        if (t > 0) { float d = hn - prevh[j]; b_acc += d * d; }  // TAR
        prevh[j] = hn;
        if (t > 0) {
          int to = dir ? (TT - t) : (t - 1);                // time-shifted concat
          out[(((size_t)to) * NB + m) * (2 * HID) + dir * HID + hbase + hh] = hn;
        }
      }
    }
    __syncthreads();
    __threadfence();                     // publish seqout[t] to all WGs
    if (tid == 0) {
      atomicAdd(&counters[t], 1);
      volatile int* vc = counters;
      while (vc[t] < NWG) __builtin_amdgcn_s_sleep(1);
    }
    __syncthreads();
    __threadfence();                     // order next step's loads after barrier
  }

  if (is_last) {
    red[tid] = a_acc;
    __syncthreads();
    for (int s = 128; s > 0; s >>= 1) {
      if (tid < s) red[tid] += red[tid + s];
      __syncthreads();
    }
    if (tid == 0) alphaP[wg] = red[0];
    __syncthreads();
    red[tid] = b_acc;
    __syncthreads();
    for (int s = 128; s > 0; s >>= 1) {
      if (tid < s) red[tid] += red[tid + s];
      __syncthreads();
    }
    if (tid == 0) betaP[wg] = red[0];
  }
}

// Deterministic fixed-order final reduction of AR/TAR partials.
__global__ void finalize_kernel(const float* __restrict__ alphaP,
                                const float* __restrict__ betaP,
                                float* __restrict__ scal) {
  if (threadIdx.x == 0 && blockIdx.x == 0) {
    float a = 0.0f, b = 0.0f;
    for (int i = 0; i < 2 * NWG; ++i) { a += alphaP[i]; b += betaP[i]; }
    scal[0] = a / (float)((size_t)TS * NB * HID);
    scal[1] = b / (float)((size_t)TT * NB * HID);
  }
}

// ------------------------------------------------------------------- launch
extern "C" void kernel_launch(void* const* d_in, const int* in_sizes, int n_in,
                              void* d_out, int out_size, void* d_ws,
                              size_t ws_size, hipStream_t stream) {
  (void)in_sizes; (void)n_in; (void)out_size; (void)ws_size;
  const float* x = (const float*)d_in[0];
  const float* W_ih[2] = {(const float*)d_in[1], (const float*)d_in[5]};
  const float* W_hh[2] = {(const float*)d_in[2], (const float*)d_in[6]};
  const float* b_ih[2] = {(const float*)d_in[3], (const float*)d_in[7]};
  const float* b_hh[2] = {(const float*)d_in[4], (const float*)d_in[8]};
  float* out = (float*)d_out;

  char* ws = (char*)d_ws;
  size_t off = 0;
  auto take = [&](size_t bytes) -> char* {
    char* p = ws + off;
    off = (off + bytes + 255) & ~(size_t)255;
    return p;
  };
  bf16_t* wih_bf = (bf16_t*)take((size_t)6 * G4 * HID * 2);   // 48 MB
  bf16_t* whh_bf = (bf16_t*)take((size_t)6 * G4 * HID * 2);   // 48 MB
  float*  bias   = (float*) take((size_t)6 * G4 * 4);
  bf16_t* seqA   = (bf16_t*)take((size_t)TS * NB * HID * 2);  // 33.6 MB
  bf16_t* seqB   = (bf16_t*)take((size_t)TS * NB * HID * 2);  // 33.6 MB
  bf16_t* xg     = (bf16_t*)take((size_t)TS * NB * G4 * 2);   // 134 MB
  int*    cnts   = (int*)   take((size_t)TS * 4);
  float*  alphaP = (float*) take((size_t)2 * NWG * 4);
  float*  betaP  = (float*) take((size_t)2 * NWG * 4);

  const int CV = 3 * G4 * HID;
  for (int d = 0; d < 2; ++d) {
    f32_to_bf16_kernel<<<(CV + 255) / 256, 256, 0, stream>>>(
        W_ih[d], wih_bf + (size_t)d * CV, CV);
    f32_to_bf16_kernel<<<(CV + 255) / 256, 256, 0, stream>>>(
        W_hh[d], whh_bf + (size_t)d * CV, CV);
    combine_bias_kernel<<<(3 * G4 + 255) / 256, 256, 0, stream>>>(
        b_ih[d], b_hh[d], bias + (size_t)d * 3 * G4, 3 * G4);
  }

  const int SEQN = TS * NB * HID;
  for (int d = 0; d < 2; ++d) {
    prep_seq_kernel<<<(SEQN + 255) / 256, 256, 0, stream>>>(x, seqA, d);
    bf16_t* sin = seqA;
    bf16_t* sout = seqB;
    for (int l = 0; l < 3; ++l) {
      const int wsel = d * 3 + l;
      gemm_xg_kernel<<<dim3(TS, 16), 256, 0, stream>>>(
          sin, wih_bf + (size_t)wsel * G4 * HID, bias + (size_t)wsel * G4, xg);
      zero_counters_kernel<<<(TS + 255) / 256, 256, 0, stream>>>(cnts, TS);
      lstm_scan_kernel<<<NWG, 256, 0, stream>>>(
          xg, whh_bf + (size_t)wsel * G4 * HID, sout, cnts, out,
          (l == 2) ? 1 : 0, d, alphaP + d * NWG, betaP + d * NWG);
      bf16_t* tmp = sin; sin = sout; sout = tmp;
    }
  }
  finalize_kernel<<<1, 1, 0, stream>>>(alphaP, betaP,
                                       out + (size_t)TT * NB * 2 * HID);
}
